// FPModule_60043642798274
// MI455X (gfx1250) — matrix operand order repaired and checked by reading, not compiled, not run on hardware
//
#include <hip/hip_runtime.h>

typedef __bf16 v16bf __attribute__((ext_vector_type(16)));
typedef __bf16 v8bf  __attribute__((ext_vector_type(8)));
typedef float  v8f   __attribute__((ext_vector_type(8)));

#define BN    8
#define NC    2048
#define NFP   8192
#define CDIM  256
#define CSDIM 64
#define OUTD  256

__device__ __forceinline__ __bf16 f2bf(float f) {
  unsigned u = __builtin_bit_cast(unsigned, f);
  u += 0x7fffu + ((u >> 16) & 1u);                 // round-to-nearest-even
  unsigned short h = (unsigned short)(u >> 16);
  return __builtin_bit_cast(__bf16, h);
}

// ---------------- elementwise f32 -> bf16 ----------------
__global__ void cvt_f32_bf16_kernel(const float* __restrict__ src,
                                    __bf16* __restrict__ dst, int n) {
  int i = blockIdx.x * blockDim.x + threadIdx.x;
  int stride = gridDim.x * blockDim.x;
  for (; i < n; i += stride) dst[i] = f2bf(src[i]);
}

// ---------------- pack W[Kdim x 256] into WMMA B-fragment order --------------
// out[((kt*16 + nt)*32 + lane)*16 + j] = W[(kt*32 + (lane<16?0:8) + (j<8?j:8+j))*256 + nt*16 + (lane&15)]
__global__ void pack_w_kernel(const float* __restrict__ W,
                              __bf16* __restrict__ out, int Kdim) {
  int total = Kdim * 256;
  int i = blockIdx.x * blockDim.x + threadIdx.x;
  if (i >= total) return;
  int j    = i & 15;
  int lane = (i >> 4) & 31;
  int nt   = (i >> 9) & 15;
  int kt   = i >> 13;
  int k = kt * 32 + ((lane < 16) ? 0 : 8) + ((j < 8) ? j : 8 + j);
  int n = nt * 16 + (lane & 15);
  out[i] = f2bf(W[k * 256 + n]);
}

// ---------------- kNN(3) per fine point ----------------
__global__ void knn_kernel(const float* __restrict__ pos,
                           const float* __restrict__ pos_skip,
                           int* __restrict__ idx_out,
                           float* __restrict__ w_out) {
  __shared__ float sp[NC * 3];                 // 24 KB of 320 KB LDS
  const int FB = NFP / 256;                    // 32 tiles per cloud
  int cloud = blockIdx.x / FB;
  int tile  = blockIdx.x % FB;
  for (int i = threadIdx.x; i < NC * 3; i += blockDim.x)
    sp[i] = pos[(size_t)cloud * NC * 3 + i];
  __syncthreads();

  int f = cloud * NFP + tile * 256 + threadIdx.x;
  float fx = pos_skip[f * 3 + 0];
  float fy = pos_skip[f * 3 + 1];
  float fz = pos_skip[f * 3 + 2];

  float b0 = 3.4e38f, b1v = 3.4e38f, b2v = 3.4e38f;
  int   i0 = 0, i1 = 0, i2 = 0;
  for (int j = 0; j < NC; ++j) {               // all lanes read same addr: LDS broadcast
    float dx = fx - sp[j * 3 + 0];
    float dy = fy - sp[j * 3 + 1];
    float dz = fz - sp[j * 3 + 2];
    float d2 = dx * dx + dy * dy + dz * dz;
    if (d2 < b2v) {
      if (d2 < b0)        { b2v = b1v; i2 = i1; b1v = b0;  i1 = i0; b0 = d2; i0 = j; }
      else if (d2 < b1v)  { b2v = b1v; i2 = i1; b1v = d2;  i1 = j; }
      else                { b2v = d2;  i2 = j; }
    }
  }
  float w0 = 1.0f / fmaxf(b0, 1e-16f);
  float w1 = 1.0f / fmaxf(b1v, 1e-16f);
  float w2 = 1.0f / fmaxf(b2v, 1e-16f);
  float inv = 1.0f / (w0 + w1 + w2);
  idx_out[f * 3 + 0] = cloud * NC + i0;  w_out[f * 3 + 0] = w0 * inv;
  idx_out[f * 3 + 1] = cloud * NC + i1;  w_out[f * 3 + 1] = w1 * inv;
  idx_out[f * 3 + 2] = cloud * NC + i2;  w_out[f * 3 + 2] = w2 * inv;
}

// -------- software-pipelined WMMA K-loop (shared by all GEMMs) --------
// acc[t] += A(16xKdim, row-major bf16, abase pre-offset by half) @ Bp tiles
template <int KDIM>
__device__ __forceinline__ void wmma_k_loop(const __bf16* __restrict__ abase,
                                            const __bf16* __restrict__ bbase,
                                            v8f acc[4]) {
  const int KT = KDIM / 32;
  union AF { v16bf v; v8bf h[2]; };
  AF af;
  v16bf bcur[4];
  af.h[0] = *(const v8bf*)(abase);
  af.h[1] = *(const v8bf*)(abase + 16);
  #pragma unroll
  for (int t = 0; t < 4; ++t)
    bcur[t] = *(const v16bf*)(bbase + t * 512);

  #pragma unroll
  for (int kt = 0; kt < KT; ++kt) {
    AF anxt;
    v16bf bnxt[4];
    if (kt + 1 < KT) {                      // issue next step's loads first
      anxt.h[0] = *(const v8bf*)(abase + (kt + 1) * 32);
      anxt.h[1] = *(const v8bf*)(abase + (kt + 1) * 32 + 16);
      #pragma unroll
      for (int t = 0; t < 4; ++t)
        bnxt[t] = *(const v16bf*)(bbase + (kt + 1) * 8192 + t * 512);
    }
    #pragma unroll
    for (int t = 0; t < 4; ++t)             // consume current buffers
      acc[t] = __builtin_amdgcn_wmma_f32_16x16x32_bf16(
          false, af.v, false, bcur[t], (short)0, acc[t], false, false);
    if (kt + 1 < KT) {                      // renamed away after full unroll
      af.v = anxt.v;
      #pragma unroll
      for (int t = 0; t < 4; ++t) bcur[t] = bnxt[t];
    }
  }
}

// ---------------- WMMA GEMM: Y1 = bf16(x) @ W1a  (16384 x 256 x 256, f32 out)
__global__ void gemm_coarse_kernel(const __bf16* __restrict__ A,
                                   const __bf16* __restrict__ Bp,
                                   float* __restrict__ Y) {
  int lane = threadIdx.x & 31;
  int wave = threadIdx.x >> 5;
  int rowM = blockIdx.x * 64 + wave * 16;
  int ct0  = blockIdx.y * 4;
  int half = (lane < 16) ? 0 : 8;
  int mA   = rowM + (lane & 15);

  const __bf16* abase = A  + (size_t)mA * CDIM + half;
  const __bf16* bbase = Bp + (size_t)lane * 16 + (size_t)ct0 * 512;

  union { v8f v; float f[8]; } acc[4];
  #pragma unroll
  for (int t = 0; t < 4; ++t)
    #pragma unroll
    for (int q = 0; q < 8; ++q) acc[t].f[q] = 0.0f;

  v8f a[4] = {acc[0].v, acc[1].v, acc[2].v, acc[3].v};
  wmma_k_loop<CDIM>(abase, bbase, a);
  #pragma unroll
  for (int t = 0; t < 4; ++t) acc[t].v = a[t];

  #pragma unroll
  for (int r = 0; r < 8; ++r) {
    int m = rowM + r + half;
    #pragma unroll
    for (int t = 0; t < 4; ++t) {
      int n = (ct0 + t) * 16 + (lane & 15);
      Y[(size_t)m * 256 + n] = acc[t].f[r];
    }
  }
}

// -------- fused: h1 = relu( interp(Y1) + bf16(x_skip) @ W1b + b1 ) -> bf16
__global__ void skip_interp_kernel(const __bf16* __restrict__ As,   // [65536 x 64] bf16
                                   const __bf16* __restrict__ Bp,   // packed W1b [64 x 256]
                                   const float* __restrict__ Y1,    // [16384 x 256] f32
                                   const int* __restrict__ idx3,
                                   const float* __restrict__ w3,
                                   const float* __restrict__ bias,
                                   __bf16* __restrict__ H1) {
  int lane = threadIdx.x & 31;
  int wave = threadIdx.x >> 5;
  int rowM = blockIdx.x * 64 + wave * 16;
  int ct0  = blockIdx.y * 4;
  int half = (lane < 16) ? 0 : 8;
  int mA   = rowM + (lane & 15);

  const __bf16* abase = As + (size_t)mA * CSDIM + half;
  const __bf16* bbase = Bp + (size_t)lane * 16 + (size_t)ct0 * 512;

  union { v8f v; float f[8]; } acc[4];
  #pragma unroll
  for (int t = 0; t < 4; ++t)
    #pragma unroll
    for (int q = 0; q < 8; ++q) acc[t].f[q] = 0.0f;

  v8f a[4] = {acc[0].v, acc[1].v, acc[2].v, acc[3].v};
  wmma_k_loop<CSDIM>(abase, bbase, a);
  #pragma unroll
  for (int t = 0; t < 4; ++t) acc[t].v = a[t];

  float bt[4];
  #pragma unroll
  for (int t = 0; t < 4; ++t) bt[t] = bias[(ct0 + t) * 16 + (lane & 15)];

  #pragma unroll
  for (int r = 0; r < 8; ++r) {
    int m = rowM + r + half;
    int base = m * 3;
    int g0 = idx3[base + 0], g1 = idx3[base + 1], g2 = idx3[base + 2];
    float ww0 = w3[base + 0], ww1 = w3[base + 1], ww2 = w3[base + 2];
    const float* y0 = Y1 + (size_t)g0 * 256;
    const float* y1 = Y1 + (size_t)g1 * 256;
    const float* y2 = Y1 + (size_t)g2 * 256;
    #pragma unroll
    for (int t = 0; t < 4; ++t) {
      int n = (ct0 + t) * 16 + (lane & 15);
      float v = acc[t].f[r] + bt[t] + ww0 * y0[n] + ww1 * y1[n] + ww2 * y2[n];
      H1[(size_t)m * 256 + n] = f2bf(fmaxf(v, 0.0f));
    }
  }
}

// ---------------- h2 = relu( bf16(h1) @ W2 + b2 ) -> f32 (d_out) -------------
__global__ void gemm2_kernel(const __bf16* __restrict__ A,
                             const __bf16* __restrict__ Bp,
                             const float* __restrict__ bias,
                             float* __restrict__ out) {
  int lane = threadIdx.x & 31;
  int wave = threadIdx.x >> 5;
  int rowM = blockIdx.x * 64 + wave * 16;
  int ct0  = blockIdx.y * 4;
  int half = (lane < 16) ? 0 : 8;
  int mA   = rowM + (lane & 15);

  const __bf16* abase = A  + (size_t)mA * OUTD + half;
  const __bf16* bbase = Bp + (size_t)lane * 16 + (size_t)ct0 * 512;

  union { v8f v; float f[8]; } acc[4];
  #pragma unroll
  for (int t = 0; t < 4; ++t)
    #pragma unroll
    for (int q = 0; q < 8; ++q) acc[t].f[q] = 0.0f;

  v8f a[4] = {acc[0].v, acc[1].v, acc[2].v, acc[3].v};
  wmma_k_loop<OUTD>(abase, bbase, a);
  #pragma unroll
  for (int t = 0; t < 4; ++t) acc[t].v = a[t];

  #pragma unroll
  for (int r = 0; r < 8; ++r) {
    int m = rowM + r + half;
    #pragma unroll
    for (int t = 0; t < 4; ++t) {
      int n = (ct0 + t) * 16 + (lane & 15);
      out[(size_t)m * 256 + n] = fmaxf(acc[t].f[r] + bias[n], 0.0f);
    }
  }
}

// ---------------- tuple tail: pos_skip, batch_skip ----------------
__global__ void tail_kernel(const float* __restrict__ pos_skip,
                            const int* __restrict__ batch_skip,
                            float* __restrict__ out, int out_size) {
  const int HN = BN * NFP * OUTD;              // 16,777,216
  const int PN = BN * NFP * 3;
  const int BT = BN * NFP;
  int i = blockIdx.x * blockDim.x + threadIdx.x;
  int stride = gridDim.x * blockDim.x;
  for (; i < PN + BT; i += stride) {
    if (i < PN) {
      int o = HN + i;
      if (o < out_size) out[o] = pos_skip[i];
    } else {
      int b = i - PN;
      int o = HN + PN + b;
      if (o < out_size) out[o] = (float)batch_skip[b];
    }
  }
}

extern "C" void kernel_launch(void* const* d_in, const int* in_sizes, int n_in,
                              void* d_out, int out_size, void* d_ws, size_t ws_size,
                              hipStream_t stream) {
  (void)in_sizes; (void)n_in; (void)ws_size;
  const float* x        = (const float*)d_in[0];
  const float* pos      = (const float*)d_in[1];
  const float* x_skip   = (const float*)d_in[3];
  const float* pos_skip = (const float*)d_in[4];
  const int*   batch_sk = (const int*)d_in[5];
  const float* W1       = (const float*)d_in[6];
  const float* b1       = (const float*)d_in[7];
  const float* W2       = (const float*)d_in[8];
  const float* b2       = (const float*)d_in[9];
  float* out = (float*)d_out;
  char*  ws  = (char*)d_ws;

  // workspace layout (256B aligned)
  size_t off = 0;
  auto alloc = [&](size_t bytes) { size_t r = off; off = (off + bytes + 255) & ~(size_t)255; return r; };
  size_t o_xb   = alloc((size_t)BN * NC  * CDIM  * 2);   // bf16 x
  size_t o_xsb  = alloc((size_t)BN * NFP * CSDIM * 2);   // bf16 x_skip
  size_t o_w1c  = alloc((size_t)CDIM  * 256 * 2);        // packed W1 coarse
  size_t o_w1s  = alloc((size_t)CSDIM * 256 * 2);        // packed W1 skip
  size_t o_w2   = alloc((size_t)OUTD  * 256 * 2);        // packed W2
  size_t o_idx  = alloc((size_t)BN * NFP * 3 * 4);
  size_t o_w3   = alloc((size_t)BN * NFP * 3 * 4);
  size_t o_y1   = alloc((size_t)BN * NC * 256 * 4);      // coarse GEMM out (f32)
  size_t o_h1   = alloc((size_t)BN * NFP * 256 * 2);     // bf16 hidden

  __bf16* xb  = (__bf16*)(ws + o_xb);
  __bf16* xsb = (__bf16*)(ws + o_xsb);
  __bf16* w1c = (__bf16*)(ws + o_w1c);
  __bf16* w1s = (__bf16*)(ws + o_w1s);
  __bf16* w2p = (__bf16*)(ws + o_w2);
  int*    idx = (int*)(ws + o_idx);
  float*  wgt = (float*)(ws + o_w3);
  float*  Y1  = (float*)(ws + o_y1);
  __bf16* H1  = (__bf16*)(ws + o_h1);

  // 1) precision conversion + weight packing
  {
    int n = BN * NC * CDIM;
    cvt_f32_bf16_kernel<<<2048, 256, 0, stream>>>(x, xb, n);
  }
  {
    int n = BN * NFP * CSDIM;
    cvt_f32_bf16_kernel<<<2048, 256, 0, stream>>>(x_skip, xsb, n);
  }
  pack_w_kernel<<<(CDIM * 256 + 255) / 256, 256, 0, stream>>>(W1, w1c, CDIM);
  pack_w_kernel<<<(CSDIM * 256 + 255) / 256, 256, 0, stream>>>(W1 + (size_t)CDIM * 256, w1s, CSDIM);
  pack_w_kernel<<<(OUTD * 256 + 255) / 256, 256, 0, stream>>>(W2, w2p, OUTD);

  // 2) kNN(3) per fine point (LDS-resident coarse cloud)
  knn_kernel<<<BN * (NFP / 256), 256, 0, stream>>>(pos, pos_skip, idx, wgt);

  // 3) coarse GEMM at 16384 rows (interp-after-GEMM refactor)
  gemm_coarse_kernel<<<dim3((BN * NC) / 64, 4), 128, 0, stream>>>(xb, w1c, Y1);

  // 4) fused skip-GEMM + interpolation + bias + relu -> bf16 h1
  skip_interp_kernel<<<dim3((BN * NFP) / 64, 4), 128, 0, stream>>>(
      xsb, w1s, Y1, idx, wgt, b1, H1);

  // 5) second MLP layer -> f32 d_out
  gemm2_kernel<<<dim3((BN * NFP) / 64, 4), 128, 0, stream>>>(H1, w2p, b2, out);

  // 6) tuple tail
  tail_kernel<<<1024, 256, 0, stream>>>(pos_skip, batch_sk, out, out_size);
}
